// DGL_GAT_1692217114862
// MI455X (gfx1250) — matrix operand order repaired and checked by reading, not compiled, not run on hardware
//
#include <hip/hip_runtime.h>

// ---------------------------------------------------------------------------
// Problem constants (match reference)
// ---------------------------------------------------------------------------
#define N_NODES   100000
#define HEADS     4
#define OUTF      32
#define HF        128          // HEADS*OUTF
#define IN_FEATS  74
#define KPAD1     96           // 74 padded up to multiple of 32
#define NGRAPHS   2048
#define PRED      128
#define NEG_SLOPE 0.2f

typedef __attribute__((ext_vector_type(16))) _Float16 v16h;
typedef __attribute__((ext_vector_type(8)))  float    v8f;

// ---------------------------------------------------------------------------
// Monotonic unsigned key for float atomic max (handles negatives)
// ---------------------------------------------------------------------------
__device__ __forceinline__ unsigned fkey(float f) {
  unsigned u = __float_as_uint(f);
  return (u & 0x80000000u) ? ~u : (u | 0x80000000u);
}
__device__ __forceinline__ float fkeyinv(unsigned k) {
  unsigned u = (k & 0x80000000u) ? (k & 0x7FFFFFFFu) : ~k;
  return __uint_as_float(u);
}

// ---------------------------------------------------------------------------
// Conversions: f32 -> f16, K-padded; weights transposed to N-major so the
// WMMA B-fragment is two contiguous 16B loads per lane.
// ---------------------------------------------------------------------------
__global__ void k_convert_feats(const float* __restrict__ in, _Float16* __restrict__ out,
                                int rows, int kin, int kpad) {
  int i = blockIdx.x * blockDim.x + threadIdx.x;
  if (i >= rows * kpad) return;
  int r = i / kpad, k = i - r * kpad;
  out[i] = (k < kin) ? (_Float16)in[(size_t)r * kin + k] : (_Float16)0.0f;
}

// W is (K x N) row-major f32 -> Wt is (N x Kpad) row-major f16, zero-padded.
__global__ void k_convert_wt(const float* __restrict__ W, _Float16* __restrict__ Wt,
                             int K, int N, int Kpad) {
  int i = blockIdx.x * blockDim.x + threadIdx.x;
  if (i >= N * Kpad) return;
  int n = i / Kpad, k = i - n * Kpad;
  Wt[i] = (k < K) ? (_Float16)W[(size_t)k * N + n] : (_Float16)0.0f;
}

// ---------------------------------------------------------------------------
// WMMA strip GEMM: C[M x 128] = A[M x K] * Bt[128 x K]^T
// (A,Bt f16 row-major, C f32). M%16==0, K%32==0 (template constant so every
// fragment load uses an immediate instruction offset -> no VALU address
// chains, load clauses, pipelined s_wait_loadcnt). One wave per 16x128 strip:
// the A fragment is loaded once per K-step and reused across 8 column tiles
// (8 f32 accumulators in VGPRs), maximizing v_wmma density per byte loaded.
// ---------------------------------------------------------------------------
union V16HU { v16h v; uint4 q[2]; };

template <int K>
__global__ void k_wmma_gemm_strip(const _Float16* __restrict__ A,
                                  const _Float16* __restrict__ Bt,
                                  float* __restrict__ C,
                                  int M) {
  int wave = (blockIdx.x * blockDim.x + threadIdx.x) >> 5;
  int lane = threadIdx.x & 31;
  if (wave * 16 >= M) return;               // wave-uniform guard (EXEC stays all-1s)

  const int half  = lane >> 4;              // 0 or 1
  const int idx16 = lane & 15;

  const _Float16* arow  = A  + (size_t)(wave * 16 + idx16) * K;   // row-major A row
  const _Float16* bbase = Bt + (size_t)idx16 * K + 16 * half;     // column of B tile 0

  v8f acc[8];
  #pragma unroll
  for (int j = 0; j < 8; ++j) acc[j] = (v8f){};

  #pragma unroll
  for (int k0 = 0; k0 < K; k0 += 32) {
    V16HU a;
    // A frag: elems 0..15 = K k0+16*half.. (contiguous)
    a.q[0] = *(const uint4*)(arow + k0 + 16 * half);
    a.q[1] = *(const uint4*)(arow + k0 + 16 * half + 8);
    #pragma unroll
    for (int j = 0; j < 8; ++j) {
      V16HU b;
      b.q[0] = *(const uint4*)(bbase + j * 16 * K + k0);
      b.q[1] = *(const uint4*)(bbase + j * 16 * K + k0 + 8);
      acc[j] = __builtin_amdgcn_wmma_f32_16x16x32_f16(false, a.v, false, b.v,
                                                      (short)0, acc[j], false, false);
    }
  }
  #pragma unroll
  for (int j = 0; j < 8; ++j) {
    float* cbase = C + (size_t)(wave * 16) * HF + j * 16 + idx16;
    #pragma unroll
    for (int v = 0; v < 8; ++v)
      cbase[(size_t)(v + 8 * half) * HF] = acc[j][v];
  }
}

// ---------------------------------------------------------------------------
// Per-node attention logits: el[n,h] = sum_f feat[n,h,f]*attn_l[h,f]
// ---------------------------------------------------------------------------
__global__ void k_elr(const float* __restrict__ feat,
                      const float* __restrict__ al, const float* __restrict__ ar,
                      float* __restrict__ el, float* __restrict__ er) {
  int i = blockIdx.x * blockDim.x + threadIdx.x;
  if (i >= N_NODES * HEADS) return;
  int n = i >> 2, h = i & 3;
  const float* f = feat + (size_t)n * HF + h * OUTF;
  float sl = 0.f, sr = 0.f;
  #pragma unroll
  for (int k = 0; k < OUTF; ++k) {
    sl += f[k] * al[h * OUTF + k];
    sr += f[k] * ar[h * OUTF + k];
  }
  el[i] = sl; er[i] = sr;
}

// ---------------------------------------------------------------------------
// Edge softmax (3 passes) + weighted aggregation
// ---------------------------------------------------------------------------
__global__ void k_edge_logits(const int* __restrict__ src, const int* __restrict__ dst,
                              const float* __restrict__ el, const float* __restrict__ er,
                              float* __restrict__ eb, unsigned* __restrict__ emaxKey, int E) {
  int i = blockIdx.x * blockDim.x + threadIdx.x;
  if (i >= E * HEADS) return;
  int e = i >> 2, h = i & 3;
  int s = src[e], d = dst[e];
  float x = el[s * HEADS + h] + er[d * HEADS + h];
  x = (x > 0.f) ? x : NEG_SLOPE * x;
  eb[i] = x;
  atomicMax(&emaxKey[d * HEADS + h], fkey(x));
}

__global__ void k_edge_exp(const int* __restrict__ dst, float* __restrict__ eb,
                           const unsigned* __restrict__ emaxKey,
                           float* __restrict__ denom, int E) {
  int i = blockIdx.x * blockDim.x + threadIdx.x;
  if (i >= E * HEADS) return;
  int e = i >> 2, h = i & 3;
  int d = dst[e];
  float m  = fkeyinv(emaxKey[d * HEADS + h]);
  float ee = __expf(eb[i] - m);
  eb[i] = ee;
  atomicAdd(&denom[d * HEADS + h], ee);
}

__global__ void k_edge_agg(const int* __restrict__ src, const int* __restrict__ dst,
                           const float* __restrict__ eb, const float* __restrict__ denom,
                           const float* __restrict__ feat, float* __restrict__ rst, int E) {
  int i = blockIdx.x * blockDim.x + threadIdx.x;
  if (i >= E * HF) return;
  int e = i >> 7, hf = i & (HF - 1), h = hf >> 5;
  int s = src[e], d = dst[e];
  float a = eb[e * HEADS + h] / denom[d * HEADS + h];
  atomicAdd(&rst[(size_t)d * HF + hf], a * feat[(size_t)s * HF + hf]);
}

// ---------------------------------------------------------------------------
// Epilogues
// ---------------------------------------------------------------------------
// Layer 1: h1 = elu(rst + res1 + bias1); also emit f16 copy for the next GEMM.
__global__ void k_post1(const float* __restrict__ rst, const float* __restrict__ res1,
                        const float* __restrict__ bias1, float* __restrict__ h1,
                        _Float16* __restrict__ h1h) {
  int i = blockIdx.x * blockDim.x + threadIdx.x;
  if (i >= N_NODES * HF) return;
  int hf = i & (HF - 1);
  float v = rst[i] + res1[i] + bias1[hf];
  v = (v > 0.f) ? v : (__expf(v) - 1.f);
  h1[i] = v;
  h1h[i] = (_Float16)v;
}

// Layer 2: node_feats = mean_h(rst + h1(residual) + bias2)
__global__ void k_post2(const float* __restrict__ rst, const float* __restrict__ h1,
                        const float* __restrict__ bias2, float* __restrict__ nf) {
  int i = blockIdx.x * blockDim.x + threadIdx.x;
  if (i >= N_NODES * OUTF) return;
  int n = i >> 5, f = i & 31;
  float s = 0.f;
  #pragma unroll
  for (int h = 0; h < HEADS; ++h) {
    int hf = h * OUTF + f;
    s += rst[(size_t)n * HF + hf] + h1[(size_t)n * HF + hf] + bias2[hf];
  }
  nf[i] = 0.25f * s;
}

__global__ void k_node_w(const float* __restrict__ nf, const float* __restrict__ ww,
                         const float* __restrict__ wb, float* __restrict__ wn) {
  int n = blockIdx.x * blockDim.x + threadIdx.x;
  if (n >= N_NODES) return;
  const float* v = nf + (size_t)n * OUTF;
  float s = wb[0];
  #pragma unroll
  for (int f = 0; f < OUTF; ++f) s += v[f] * ww[f];
  wn[n] = 1.f / (1.f + __expf(-s));
}

__global__ void k_scatter_graph(const float* __restrict__ nf, const float* __restrict__ wn,
                                const int* __restrict__ gids, float* __restrict__ gsum,
                                unsigned* __restrict__ gmaxKey) {
  int i = blockIdx.x * blockDim.x + threadIdx.x;
  if (i >= N_NODES * OUTF) return;
  int n = i >> 5, f = i & 31;
  int g = gids[n];
  float v = nf[i];
  atomicAdd(&gsum[g * OUTF + f], wn[n] * v);
  atomicMax(&gmaxKey[g * OUTF + f], fkey(v));
}

// out[b,p] = sum_f gsum[b,f]*tw[f,p] + gmax[b,f]*tw[32+f,p] + tb[p]
__global__ void k_final(const float* __restrict__ gsum, const unsigned* __restrict__ gmaxKey,
                        const float* __restrict__ tw, const float* __restrict__ tb,
                        float* __restrict__ out) {
  int i = blockIdx.x * blockDim.x + threadIdx.x;
  if (i >= NGRAPHS * PRED) return;
  int b = i >> 7, p = i & (PRED - 1);
  float acc = tb[p];
  #pragma unroll 4
  for (int f = 0; f < OUTF; ++f) {
    acc += gsum[b * OUTF + f] * tw[f * PRED + p];
    acc += fkeyinv(gmaxKey[b * OUTF + f]) * tw[(OUTF + f) * PRED + p];
  }
  out[i] = acc;
}

// ---------------------------------------------------------------------------
// Host-side orchestration
// ---------------------------------------------------------------------------
static inline int grid1(long long n, int bs = 256) { return (int)((n + bs - 1) / bs); }

extern "C" void kernel_launch(void* const* d_in, const int* in_sizes, int n_in,
                              void* d_out, int out_size, void* d_ws, size_t ws_size,
                              hipStream_t stream) {
  const float* feats   = (const float*)d_in[0];
  const int*   src     = (const int*)  d_in[1];
  const int*   dst     = (const int*)  d_in[2];
  const int*   gids    = (const int*)  d_in[3];
  const float* fc1_w   = (const float*)d_in[4];
  const float* attn_l1 = (const float*)d_in[5];
  const float* attn_r1 = (const float*)d_in[6];
  const float* res1_w  = (const float*)d_in[7];
  const float* bias1   = (const float*)d_in[8];
  const float* fc2_w   = (const float*)d_in[9];
  const float* attn_l2 = (const float*)d_in[10];
  const float* attn_r2 = (const float*)d_in[11];
  const float* bias2   = (const float*)d_in[12];
  const float* ww      = (const float*)d_in[13];
  const float* wb      = (const float*)d_in[14];
  const float* tw      = (const float*)d_in[15];
  const float* tb      = (const float*)d_in[16];
  float* out = (float*)d_out;

  const int E = in_sizes[1];                 // 900,000 edges (incl. self-loops)
  const int N = N_NODES;

  // ---- carve workspace ----
  char* base = (char*)d_ws;
  size_t off = 0;
  auto carve = [&](size_t bytes) -> void* {
    void* r = base + off;
    off += (bytes + 255) & ~(size_t)255;
    return r;
  };
  _Float16* featsH = (_Float16*)carve((size_t)N * KPAD1 * 2);
  _Float16* w1t    = (_Float16*)carve((size_t)HF * KPAD1 * 2);
  _Float16* r1t    = (_Float16*)carve((size_t)HF * KPAD1 * 2);
  _Float16* w2t    = (_Float16*)carve((size_t)HF * HF * 2);
  float*    feat1  = (float*)   carve((size_t)N * HF * 4);  // reused as feat2
  float*    res1   = (float*)   carve((size_t)N * HF * 4);
  float*    rst    = (float*)   carve((size_t)N * HF * 4);
  float*    h1     = (float*)   carve((size_t)N * HF * 4);
  _Float16* h1h    = (_Float16*)carve((size_t)N * HF * 2);
  float*    el     = (float*)   carve((size_t)N * HEADS * 4);
  float*    er     = (float*)   carve((size_t)N * HEADS * 4);
  float*    eb     = (float*)   carve((size_t)E * HEADS * 4);
  unsigned* emaxK  = (unsigned*)carve((size_t)N * HEADS * 4);
  float*    denom  = (float*)   carve((size_t)N * HEADS * 4);
  float*    nf     = (float*)   carve((size_t)N * OUTF * 4);
  float*    wn     = (float*)   carve((size_t)N * 4);
  float*    gsum   = (float*)   carve((size_t)NGRAPHS * OUTF * 4);
  unsigned* gmaxK  = (unsigned*)carve((size_t)NGRAPHS * OUTF * 4);
  (void)ws_size;

  // ---- f16 conversions / weight transposes ----
  k_convert_feats<<<grid1((long long)N * KPAD1), 256, 0, stream>>>(feats, featsH, N, IN_FEATS, KPAD1);
  k_convert_wt  <<<grid1((long long)HF * KPAD1), 256, 0, stream>>>(fc1_w,  w1t, IN_FEATS, HF, KPAD1);
  k_convert_wt  <<<grid1((long long)HF * KPAD1), 256, 0, stream>>>(res1_w, r1t, IN_FEATS, HF, KPAD1);
  k_convert_wt  <<<grid1((long long)HF * HF),    256, 0, stream>>>(fc2_w,  w2t, HF, HF, HF);

  // One wave per 16-row strip; 6250 strips for N_NODES=100000.
  const long long stripWaves  = (long long)N / 16;
  const int       stripBlocks = grid1(stripWaves * 32);

  // ---- Layer 1 GEMMs (WMMA): feat1 = feats@fc1, res1 = feats@res1_w ----
  k_wmma_gemm_strip<KPAD1><<<stripBlocks, 256, 0, stream>>>(featsH, w1t, feat1, N);
  k_wmma_gemm_strip<KPAD1><<<stripBlocks, 256, 0, stream>>>(featsH, r1t, res1,  N);

  // ---- Layer 1 attention ----
  k_elr<<<grid1((long long)N * HEADS), 256, 0, stream>>>(feat1, attn_l1, attn_r1, el, er);
  hipMemsetAsync(emaxK, 0, (size_t)N * HEADS * 4, stream);
  hipMemsetAsync(denom, 0, (size_t)N * HEADS * 4, stream);
  k_edge_logits<<<grid1((long long)E * HEADS), 256, 0, stream>>>(src, dst, el, er, eb, emaxK, E);
  k_edge_exp   <<<grid1((long long)E * HEADS), 256, 0, stream>>>(dst, eb, emaxK, denom, E);
  hipMemsetAsync(rst, 0, (size_t)N * HF * 4, stream);
  k_edge_agg   <<<grid1((long long)E * HF),    256, 0, stream>>>(src, dst, eb, denom, feat1, rst, E);
  k_post1      <<<grid1((long long)N * HF),    256, 0, stream>>>(rst, res1, bias1, h1, h1h);

  // ---- Layer 2 GEMM (WMMA): feat2 = h1@fc2 (reuse feat1 buffer) ----
  k_wmma_gemm_strip<HF><<<stripBlocks, 256, 0, stream>>>(h1h, w2t, feat1, N);

  // ---- Layer 2 attention ----
  k_elr<<<grid1((long long)N * HEADS), 256, 0, stream>>>(feat1, attn_l2, attn_r2, el, er);
  hipMemsetAsync(emaxK, 0, (size_t)N * HEADS * 4, stream);
  hipMemsetAsync(denom, 0, (size_t)N * HEADS * 4, stream);
  k_edge_logits<<<grid1((long long)E * HEADS), 256, 0, stream>>>(src, dst, el, er, eb, emaxK, E);
  k_edge_exp   <<<grid1((long long)E * HEADS), 256, 0, stream>>>(dst, eb, emaxK, denom, E);
  hipMemsetAsync(rst, 0, (size_t)N * HF * 4, stream);
  k_edge_agg   <<<grid1((long long)E * HF),    256, 0, stream>>>(src, dst, eb, denom, feat1, rst, E);
  k_post2      <<<grid1((long long)N * OUTF),  256, 0, stream>>>(rst, h1, bias2, nf);

  // ---- Readout ----
  k_node_w<<<grid1((long long)N), 256, 0, stream>>>(nf, ww, wb, wn);
  hipMemsetAsync(gsum,  0, (size_t)NGRAPHS * OUTF * 4, stream);
  hipMemsetAsync(gmaxK, 0, (size_t)NGRAPHS * OUTF * 4, stream);
  k_scatter_graph<<<grid1((long long)N * OUTF), 256, 0, stream>>>(nf, wn, gids, gsum, gmaxK);
  k_final<<<grid1((long long)NGRAPHS * PRED), 256, 0, stream>>>(gsum, gmaxK, tw, tb, out);
  (void)n_in; (void)out_size;
}
